// ScaledDotProductAttention_42451456754057
// MI455X (gfx1250) — matrix operand order repaired and checked by reading, not compiled, not run on hardware
//
#include <hip/hip_runtime.h>

typedef __attribute__((ext_vector_type(16))) _Float16 v16h;
typedef __attribute__((ext_vector_type(8)))  _Float16 v8h;
typedef __attribute__((ext_vector_type(4)))  _Float16 v4h;
typedef __attribute__((ext_vector_type(8)))  float    v8f;
typedef __attribute__((ext_vector_type(4)))  float    v4f;

#define WAVES 8
#define BM    128   // queries per workgroup (16 per wave)
#define BN    64    // keys per iteration
#define DHEAD 128
#define NQ    2048
#define NK    2048

static __device__ __forceinline__ v16h cat16(v8h lo, v8h hi) {
    return __builtin_shufflevector(lo, hi, 0,1,2,3,4,5,6,7,8,9,10,11,12,13,14,15);
}
static __device__ __forceinline__ v4h cvt4(v4f f) {
    v4h h;
    #pragma unroll
    for (int j = 0; j < 4; ++j) h[j] = (_Float16)f[j];
    return h;
}

__global__ __launch_bounds__(256)
void fa2_wmma_kernel(const float* __restrict__ Q, const float* __restrict__ K,
                     const float* __restrict__ V, const int* __restrict__ vlen_p,
                     float* __restrict__ O)
{
    // Double-buffered K tile row-major [key][d] and V tile transposed [d][key].
    // P is per-wave scratch for the C-layout -> A-layout round trip.
    __shared__ __align__(16) _Float16 Kt[2][BN][DHEAD + 8];       // 2 x 64 x 136
    __shared__ __align__(16) _Float16 Vt[2][DHEAD][BN + 8];       // 2 x 128 x 72
    __shared__ __align__(16) _Float16 Pt[WAVES][16][BN + 8];      // 8 x 16 x 72

    const int tid  = threadIdx.x;
    const int lane = tid & 31;
    const int wave = tid >> 5;
    const int half = lane >> 4;     // 0: lanes 0-15, 1: lanes 16-31
    const int l16  = lane & 15;
    const int b    = blockIdx.y;
    const int q0   = blockIdx.x * BM + wave * 16;

    const int   vlen  = vlen_p[b];
    const float scale = 0.08838834764831845f;   // 1/sqrt(128)

    // ---- Load Q (16 rows x 128) into f16 A-fragments, once.
    v16h qfrag[4];
    {
        const float* qrow = Q + ((size_t)b * NQ + (size_t)(q0 + l16)) * DHEAD;
        const int kb = half * 8;
        #pragma unroll
        for (int s = 0; s < 4; ++s) {
            v8h chunk[2];
            #pragma unroll
            for (int c = 0; c < 2; ++c) {
                const float* src = qrow + s * 32 + c * 16 + kb;
                v4f f0 = *(const v4f*)(src);
                v4f f1 = *(const v4f*)(src + 4);
                v8h h;
                #pragma unroll
                for (int j = 0; j < 4; ++j) {
                    h[j]     = (_Float16)f0[j];
                    h[4 + j] = (_Float16)f1[j];
                }
                chunk[c] = h;
            }
            qfrag[s] = cat16(chunk[0], chunk[1]);
        }
    }

    // ---- Online-softmax state (C-frag rows: r -> row half*8+r, col l16)
    float m[8], l[8];
    v8f   acc[8];
    #pragma unroll
    for (int r = 0; r < 8; ++r) { m[r] = -3.0e38f; l[r] = 0.0f; }
    #pragma unroll
    for (int n = 0; n < 8; ++n) {
        v8f z = {0.f,0.f,0.f,0.f,0.f,0.f,0.f,0.f};
        acc[n] = z;
    }

    const int nblk = (vlen + BN - 1) / BN;   // skip fully-masked KV blocks (exp underflows to 0)

    // ---- Prologue: stage tile 0 into buffer 0 (f16 conversion before the store)
    {
        v4h kh[8], vh[8];
        #pragma unroll
        for (int i = 0; i < 8; ++i) {
            const int idx = tid + i * 256;
            const int row = idx >> 5, c4 = idx & 31;
            const size_t base = ((size_t)b * NK + (size_t)row) * DHEAD + c4 * 4;
            kh[i] = cvt4(*(const v4f*)(K + base));
            vh[i] = cvt4(*(const v4f*)(V + base));
        }
        #pragma unroll
        for (int i = 0; i < 8; ++i) {
            const int idx = tid + i * 256;
            const int row = idx >> 5, c4 = idx & 31;
            *(v4h*)&Kt[0][row][c4 * 4] = kh[i];
            #pragma unroll
            for (int j = 0; j < 4; ++j) Vt[0][c4 * 4 + j][row] = vh[i][j];
        }
    }

    for (int kblk = 0; kblk < nblk; ++kblk) {
        const int bufc    = kblk & 1;
        const int bufn    = bufc ^ 1;
        const int kvbase  = kblk * BN;
        const bool hasNext = (kblk + 1) < nblk;

        // ---- Prefetch NEXT tile: global load + immediate f16 pack (32 VGPRs held, not 64)
        v4h kh[8], vh[8];
        if (hasNext) {
            const int kvnext = kvbase + BN;
            v4f kf[8], vf[8];
            #pragma unroll
            for (int i = 0; i < 8; ++i) {
                const int idx = tid + i * 256;
                const int row = idx >> 5, c4 = idx & 31;
                const size_t base = ((size_t)b * NK + (size_t)(kvnext + row)) * DHEAD + c4 * 4;
                kf[i] = *(const v4f*)(K + base);
                vf[i] = *(const v4f*)(V + base);
            }
            #pragma unroll
            for (int i = 0; i < 8; ++i) { kh[i] = cvt4(kf[i]); vh[i] = cvt4(vf[i]); }
        }

        // Single barrier per iteration:
        //  (a) tile kblk's stores (issued last iteration) are complete in buf[bufc]
        //  (b) every wave is done reading buf[bufn] from iteration kblk-1
        __syncthreads();

        // ---- Drain next tile into the alternate buffer (overlaps WMMAs via in-order DS pipe)
        if (hasNext) {
            #pragma unroll
            for (int i = 0; i < 8; ++i) {
                const int idx = tid + i * 256;
                const int row = idx >> 5, c4 = idx & 31;
                *(v4h*)&Kt[bufn][row][c4 * 4] = kh[i];
                #pragma unroll
                for (int j = 0; j < 4; ++j) Vt[bufn][c4 * 4 + j][row] = vh[i][j];
            }
        }

        // ---- S = Q @ K^T : s outer / t inner so consecutive WMMAs hit different accumulators;
        //      2-deep fragment rotation issues the next B load before the current WMMA.
        v8f sc[4];
        #pragma unroll
        for (int s = 0; s < 4; ++s) {
            const _Float16* bp0 = &Kt[bufc][l16][s * 32 + half * 8];
            v16h bcur = cat16(*(const v8h*)bp0, *(const v8h*)(bp0 + 16));
            #pragma unroll
            for (int t = 0; t < 4; ++t) {
                v16h bnxt = bcur;
                if (t < 3) {
                    const _Float16* bp = &Kt[bufc][(t + 1) * 16 + l16][s * 32 + half * 8];
                    bnxt = cat16(*(const v8h*)bp, *(const v8h*)(bp + 16));
                }
                if (s == 0) {
                    v8f z = {0.f,0.f,0.f,0.f,0.f,0.f,0.f,0.f};
                    sc[t] = __builtin_amdgcn_wmma_f32_16x16x32_f16(false, qfrag[s], false, bcur,
                                                                   (short)0, z, false, false);
                } else {
                    sc[t] = __builtin_amdgcn_wmma_f32_16x16x32_f16(false, qfrag[s], false, bcur,
                                                                   (short)0, sc[t], false, false);
                }
                bcur = bnxt;
            }
        }

        // ---- scale + mask (column fixed per lane per tile)
        #pragma unroll
        for (int t = 0; t < 4; ++t) {
            const bool oob = (kvbase + t * 16 + l16) >= vlen;
            #pragma unroll
            for (int r = 0; r < 8; ++r) {
                float s = sc[t][r] * scale;
                sc[t][r] = oob ? -1.0e6f : s;
            }
        }

        // ---- online softmax
        float mnew[8], fac[8];
        #pragma unroll
        for (int r = 0; r < 8; ++r) {
            float mx = fmaxf(fmaxf(sc[0][r], sc[1][r]), fmaxf(sc[2][r], sc[3][r]));
            mx = fmaxf(mx, __shfl_xor(mx, 1));
            mx = fmaxf(mx, __shfl_xor(mx, 2));
            mx = fmaxf(mx, __shfl_xor(mx, 4));
            mx = fmaxf(mx, __shfl_xor(mx, 8));
            mnew[r] = fmaxf(m[r], mx);
            fac[r]  = __expf(m[r] - mnew[r]);
            m[r]    = mnew[r];
        }
        #pragma unroll
        for (int r = 0; r < 8; ++r) {
            float sum = 0.0f;
            #pragma unroll
            for (int t = 0; t < 4; ++t) {
                float p = __expf(sc[t][r] - mnew[r]);
                sc[t][r] = p;
                sum += p;
            }
            sum += __shfl_xor(sum, 1);
            sum += __shfl_xor(sum, 2);
            sum += __shfl_xor(sum, 4);
            sum += __shfl_xor(sum, 8);
            l[r] = l[r] * fac[r] + sum;
        }
        #pragma unroll
        for (int n = 0; n < 8; ++n)
            #pragma unroll
            for (int r = 0; r < 8; ++r)
                acc[n][r] *= fac[r];

        // ---- P: C-layout -> per-wave LDS -> A-fragments
        #pragma unroll
        for (int t = 0; t < 4; ++t)
            #pragma unroll
            for (int r = 0; r < 8; ++r)
                Pt[wave][half * 8 + r][t * 16 + l16] = (_Float16)sc[t][r];

        v16h pfrag[2];
        {
            const int kb = half * 8;
            #pragma unroll
            for (int s = 0; s < 2; ++s) {
                const _Float16* pp = &Pt[wave][l16][s * 32 + kb];
                pfrag[s] = cat16(*(const v8h*)pp, *(const v8h*)(pp + 16));
            }
        }

        // ---- O += P @ V : s outer / n inner (8 independent accumulators), 2-deep rotation
        #pragma unroll
        for (int s = 0; s < 2; ++s) {
            const _Float16* vp0 = &Vt[bufc][l16][s * 32 + half * 8];
            v16h bcur = cat16(*(const v8h*)vp0, *(const v8h*)(vp0 + 16));
            #pragma unroll
            for (int n = 0; n < 8; ++n) {
                v16h bnxt = bcur;
                if (n < 7) {
                    const _Float16* vp = &Vt[bufc][(n + 1) * 16 + l16][s * 32 + half * 8];
                    bnxt = cat16(*(const v8h*)vp, *(const v8h*)(vp + 16));
                }
                acc[n] = __builtin_amdgcn_wmma_f32_16x16x32_f16(false, pfrag[s], false, bcur,
                                                                (short)0, acc[n], false, false);
                bcur = bnxt;
            }
        }
    }

    // ---- epilogue: normalize and store fp32
    float inv[8];
    #pragma unroll
    for (int r = 0; r < 8; ++r) inv[r] = 1.0f / l[r];
    float* orow = O + ((size_t)b * NQ + (size_t)q0) * DHEAD;
    #pragma unroll
    for (int n = 0; n < 8; ++n)
        #pragma unroll
        for (int r = 0; r < 8; ++r)
            orow[(size_t)(half * 8 + r) * DHEAD + n * 16 + l16] = acc[n][r] * inv[r];
}

extern "C" void kernel_launch(void* const* d_in, const int* in_sizes, int n_in,
                              void* d_out, int out_size, void* d_ws, size_t ws_size,
                              hipStream_t stream) {
    const float* q    = (const float*)d_in[0];
    const float* k    = (const float*)d_in[1];
    const float* v    = (const float*)d_in[2];
    const int*   vlen = (const int*)d_in[3];
    float*       out  = (float*)d_out;

    const int B = in_sizes[3];                 // 32
    dim3 grid(NQ / BM, B);
    dim3 block(256);
    fa2_wmma_kernel<<<grid, block, 0, stream>>>(q, k, v, vlen, out);
}
